// MultiHeadAttention_45157286150699
// MI455X (gfx1250) — matrix operand order repaired and checked by reading, not compiled, not run on hardware
//
#include <hip/hip_runtime.h>

// ---------------------------------------------------------------------------
// MI455X (gfx1250) fused multi-head attention forward, bf16 WMMA pipeline.
//   1) gemm_wmma<float,OUT_HEADS>    : Q = x @ Wq^T + bq  -> [B,H,L,64] bf16
//   2) gemm_wmma<float,OUT_HEADS>    : K = x @ Wk^T + bk  -> [B,H,L,64] bf16
//   3) gemm_wmma<float,OUT_HEADS_T>  : V = x @ Wv^T + bv  -> [B,H,64,L] bf16
//   4) flash_attn                    : streaming softmax(QK^T/8) V -> [B,L,1024] bf16
//   5) gemm_wmma<bf16,OUT_F32>       : out = attn @ Wo^T + bo -> [B,L,1024] f32
// Async staging (ASYNCcnt) for all bf16->bf16 tile copies.
// ---------------------------------------------------------------------------

typedef __attribute__((ext_vector_type(16))) __bf16 v16bf;
typedef __attribute__((ext_vector_type(8)))  float  v8f;

#define DEVINL __device__ __forceinline__

union FragAB {           // 16x32 (A) or 32x16 (B) bf16 operand: 8 VGPRs
  v16bf f;
  uint4 u[2];
};

DEVINL v8f vzero8() {
  v8f z;
#pragma unroll
  for (int i = 0; i < 8; ++i) z[i] = 0.0f;
  return z;
}

DEVINL v8f wmma_bf16(v16bf a, v16bf b, v8f c) {
  // (neg_a, A, neg_b, B, c_mod, C, reuse_a, reuse_b)
  return __builtin_amdgcn_wmma_f32_16x16x32_bf16(false, a, false, b,
                                                 (short)0, c, false, false);
}

// Truncate generic pointer to the 32-bit LDS byte offset (flat addr[31:0]).
DEVINL unsigned lds_addr32(const void* p) { return (unsigned)(size_t)p; }

// Per-lane async copy of 32 contiguous bytes global -> LDS (ASYNCcnt-tracked).
DEVINL void async_copy32(unsigned lds_dst, const void* gsrc) {
  asm volatile(
      "global_load_async_to_lds_b128 %0, %1, off\n\t"
      "global_load_async_to_lds_b128 %0, %1, off offset:16"
      :
      : "v"(lds_dst), "v"(gsrc)
      : "memory");
}

// Per-lane async copy of 64 contiguous bytes global -> LDS.
DEVINL void async_copy64(unsigned lds_dst, const void* gsrc) {
  asm volatile(
      "global_load_async_to_lds_b128 %0, %1, off\n\t"
      "global_load_async_to_lds_b128 %0, %1, off offset:16\n\t"
      "global_load_async_to_lds_b128 %0, %1, off offset:32\n\t"
      "global_load_async_to_lds_b128 %0, %1, off offset:48"
      :
      : "v"(lds_dst), "v"(gsrc)
      : "memory");
}

DEVINL void wait_async0() {
  asm volatile("s_wait_asynccnt 0x0" ::: "memory");
}

// A-matrix 16x32 bf16 fragment (ISA 7.12.2): lane m = lane&15, hs = lane>>4.
// VGPR0-3 hold K = hs*8 .. hs*8+7, VGPR4-7 hold K = 16+hs*8 .. 16+hs*8+7.
DEVINL void load_fragA(FragAB& fr, const __bf16* base, int ld) {
  const int lane = threadIdx.x & 31;
  const __bf16* p = base + (size_t)(lane & 15) * ld + ((lane >> 4) << 3);
  fr.u[0] = *(const uint4*)p;
  fr.u[1] = *(const uint4*)(p + 16);
}

// B-matrix 32x16 bf16 fragment, operand supplied as B^T rows (N x K row-major).
// lane n = lane&15, hs = lane>>4; VGPR0-3 hold K = hs*8 .. hs*8+7,
// VGPR4-7 hold K = 16+hs*8 .. 16+hs*8+7 (same order as A fragment).
DEVINL void load_fragB(FragAB& fr, const __bf16* base, int ld) {
  const int lane = threadIdx.x & 31;
  const __bf16* p = base + (size_t)(lane & 15) * ld + ((lane >> 4) << 3);
  fr.u[0] = *(const uint4*)p;
  fr.u[1] = *(const uint4*)(p + 16);
}

// ---------------------------------------------------------------------------
constexpr int DM = 1024;     // d_model
constexpr int LQ = 2048;     // sequence length
constexpr int NH = 16;       // heads
constexpr int DKH = 64;      // head dim
constexpr int MROWS = 8192;  // B*L

constexpr int BM = 128, BN = 64, KC = 32;
constexpr int LDA = KC + 8;  // padded LDS stride (80B rows, 16B aligned)

enum { OUT_F32 = 0, OUT_HEADS = 1, OUT_HEADS_T = 2 };

// Y[m,n] = sum_k A[m,k] * W[n,k] + bias[n]; A f32 or bf16, W f32 (cast in staging)
template <typename AT, int MODE>
__global__ __launch_bounds__(128) void gemm_wmma(const AT* __restrict__ A,
                                                 const float* __restrict__ W,
                                                 const float* __restrict__ bias,
                                                 void* __restrict__ outp) {
  __shared__ __align__(16) __bf16 As[BM * LDA];
  __shared__ __align__(16) __bf16 Ws[BN * LDA];

  const int tid = threadIdx.x;
  const int wave = tid >> 5, lane = tid & 31;
  const int r15 = lane & 15, hs = lane >> 4;
  const int wgM = blockIdx.x * BM, wgN = blockIdx.y * BN;

  v8f acc[2][4];
#pragma unroll
  for (int i = 0; i < 2; ++i)
#pragma unroll
    for (int j = 0; j < 4; ++j) acc[i][j] = vzero8();

  for (int k0 = 0; k0 < DM; k0 += KC) {
    __syncthreads();
    // ---- stage A tile: 128 rows x 32 K, one row per thread ----
    if constexpr (__is_same(AT, float)) {  // f32 -> bf16 through registers
      union { __bf16 h[KC]; uint4 q[4]; } cvt;
      const float4* s4 = (const float4*)(A + (size_t)(wgM + tid) * DM + k0);
#pragma unroll
      for (int c = 0; c < 8; ++c) {
        float4 t = s4[c];
        cvt.h[4 * c + 0] = (__bf16)t.x;
        cvt.h[4 * c + 1] = (__bf16)t.y;
        cvt.h[4 * c + 2] = (__bf16)t.z;
        cvt.h[4 * c + 3] = (__bf16)t.w;
      }
      uint4* dst = (uint4*)&As[tid * LDA];
#pragma unroll
      for (int c = 0; c < 4; ++c) dst[c] = cvt.q[c];
    } else {  // bf16 -> bf16: async DMA straight into LDS
      async_copy64(lds_addr32(&As[tid * LDA]),
                   A + (size_t)(wgM + tid) * DM + k0);
    }
    // ---- stage W tile: 64 rows x 32 K, half-row (16) per thread ----
    {
      const int wr = tid & 63, wh = tid >> 6;
      union { __bf16 h[16]; uint4 q[2]; } cvt;
      const float4* s4 =
          (const float4*)(W + (size_t)(wgN + wr) * DM + k0 + wh * 16);
#pragma unroll
      for (int c = 0; c < 4; ++c) {
        float4 t = s4[c];
        cvt.h[4 * c + 0] = (__bf16)t.x;
        cvt.h[4 * c + 1] = (__bf16)t.y;
        cvt.h[4 * c + 2] = (__bf16)t.z;
        cvt.h[4 * c + 3] = (__bf16)t.w;
      }
      uint4* dst = (uint4*)&Ws[wr * LDA + wh * 16];
      dst[0] = cvt.q[0];
      dst[1] = cvt.q[1];
    }
    if (k0 + KC < DM) {  // global_prefetch_b8 for next K slab
      __builtin_prefetch(A + (size_t)(wgM + tid) * DM + k0 + KC, 0, 1);
      __builtin_prefetch(W + (size_t)(wgN + (tid & 63)) * DM + k0 + KC, 0, 1);
    }
    if constexpr (!__is_same(AT, float)) wait_async0();
    __syncthreads();

    FragAB aF0, aF1, bF;
    load_fragA(aF0, &As[(wave * 32 + 0) * LDA], LDA);
    load_fragA(aF1, &As[(wave * 32 + 16) * LDA], LDA);
#pragma unroll
    for (int j = 0; j < 4; ++j) {
      load_fragB(bF, &Ws[(j * 16) * LDA], LDA);
      acc[0][j] = wmma_bf16(aF0.f, bF.f, acc[0][j]);
      acc[1][j] = wmma_bf16(aF1.f, bF.f, acc[1][j]);
    }
  }

  // ---- epilogue: C layout row m = vgpr + 8*hs, col n = lane&15 ----
#pragma unroll
  for (int j = 0; j < 4; ++j) {
    const int n = wgN + j * 16 + r15;
    const float bn = bias[n];
#pragma unroll
    for (int i = 0; i < 2; ++i) {
#pragma unroll
      for (int v = 0; v < 8; ++v) {
        const int m = wgM + wave * 32 + i * 16 + v + hs * 8;
        const float val = acc[i][j][v] + bn;
        if constexpr (MODE == OUT_F32) {
          ((float*)outp)[(size_t)m * DM + n] = val;
        } else {
          const int b = m >> 11, l = m & (LQ - 1);
          const int h = n >> 6, d = n & (DKH - 1);
          size_t idx;
          if constexpr (MODE == OUT_HEADS)
            idx = (((size_t)(b * NH + h)) * LQ + l) * DKH + d;
          else
            idx = (((size_t)(b * NH + h)) * DKH + d) * LQ + l;
          ((__bf16*)outp)[idx] = (__bf16)val;
        }
      }
    }
  }
}

// ---------------------------------------------------------------------------
// Flash attention: block = 4 waves = 64 Q rows of one (b,h); stream K in 64s.
constexpr int BQ = 64, BK = 64, FLD = 72;  // padded LDS stride (144B rows)

__global__ __launch_bounds__(128) void flash_attn(const __bf16* __restrict__ Qp,
                                                  const __bf16* __restrict__ Kp,
                                                  const __bf16* __restrict__ Vt,
                                                  __bf16* __restrict__ AO) {
  __shared__ __align__(16) __bf16 Ks[BK * FLD];      // [k][d]
  __shared__ __align__(16) __bf16 Vs[DKH * FLD];     // [d][k]  (from V^T)
  __shared__ __align__(16) __bf16 Ps[4 * 16 * FLD];  // per-wave P tile [q][k]

  const int tid = threadIdx.x, wave = tid >> 5, lane = tid & 31;
  const int r15 = lane & 15, hs = lane >> 4;

  const int bh = blockIdx.x >> 5;  // (b*16+h), 32 q-blocks per head
  const int qblk = blockIdx.x & 31;
  const int b = bh >> 4, h = bh & 15;
  const size_t hbase = (size_t)bh * LQ * DKH;  // Qp / Kp head base
  const size_t vbase = (size_t)bh * DKH * LQ;  // Vt head base
  const int q0 = qblk * BQ + wave * 16;

  FragAB qF[2];
  load_fragA(qF[0], Qp + hbase + (size_t)q0 * DKH + 0, DKH);
  load_fragA(qF[1], Qp + hbase + (size_t)q0 * DKH + 32, DKH);

  v8f o[4];
#pragma unroll
  for (int dt = 0; dt < 4; ++dt) o[dt] = vzero8();
  float mrow[8], lrow[8];
#pragma unroll
  for (int v = 0; v < 8; ++v) { mrow[v] = -1.0e30f; lrow[v] = 0.0f; }

  const float sc = 0.125f * 1.44269504088896f;  // 1/sqrt(64) * log2(e)

  for (int kp = 0; kp < LQ; kp += BK) {
    __syncthreads();
    {  // async-stage K (64x64) and V^T (64x64): 32B (16 elems) at col hf*32 per thread
      const int r = tid & 63, hf = tid >> 6;
      async_copy32(lds_addr32(&Ks[r * FLD + hf * 32]),
                   Kp + hbase + (size_t)(kp + r) * DKH + hf * 32);
      async_copy32(lds_addr32(&Vs[r * FLD + hf * 32]),
                   Vt + vbase + (size_t)r * LQ + kp + hf * 32);
    }
    wait_async0();
    __syncthreads();

    // S = Q K^T  (then scaled into log2 domain)
    v8f s[4];
#pragma unroll
    for (int jt = 0; jt < 4; ++jt) s[jt] = vzero8();
#pragma unroll
    for (int jt = 0; jt < 4; ++jt) {
      FragAB kF;
#pragma unroll
      for (int c = 0; c < 2; ++c) {
        load_fragB(kF, &Ks[(jt * 16) * FLD + c * 32], FLD);
        s[jt] = wmma_bf16(qF[c].f, kF.f, s[jt]);
      }
    }
#pragma unroll
    for (int jt = 0; jt < 4; ++jt)
#pragma unroll
      for (int v = 0; v < 8; ++v) s[jt][v] *= sc;

    // online softmax: rows live at (vgpr v, lane-half); reduce over 16 lanes
    float mnew[8], corr[8];
#pragma unroll
    for (int v = 0; v < 8; ++v) {
      float mx = fmaxf(fmaxf(s[0][v], s[1][v]), fmaxf(s[2][v], s[3][v]));
#pragma unroll
      for (int off = 1; off < 16; off <<= 1)
        mx = fmaxf(mx, __shfl_xor(mx, off, 32));
      mnew[v] = fmaxf(mrow[v], mx);
      corr[v] = exp2f(mrow[v] - mnew[v]);
      mrow[v] = mnew[v];
    }
#pragma unroll
    for (int v = 0; v < 8; ++v) {
      float rs = 0.0f;
#pragma unroll
      for (int jt = 0; jt < 4; ++jt) {
        const float p = exp2f(s[jt][v] - mnew[v]);
        s[jt][v] = p;
        rs += p;
      }
#pragma unroll
      for (int off = 1; off < 16; off <<= 1) rs += __shfl_xor(rs, off, 32);
      lrow[v] = lrow[v] * corr[v] + rs;
#pragma unroll
      for (int dt = 0; dt < 4; ++dt) o[dt][v] *= corr[v];
    }

    // P: C-layout regs -> LDS row-major -> A fragments (per-wave private tile)
    __bf16* Pw = &Ps[wave * 16 * FLD];
#pragma unroll
    for (int v = 0; v < 8; ++v) {
      const int pr = v + hs * 8;
#pragma unroll
      for (int jt = 0; jt < 4; ++jt)
        Pw[pr * FLD + jt * 16 + r15] = (__bf16)s[jt][v];
    }
    asm volatile("s_wait_dscnt 0x0" ::: "memory");  // wave-local LDS RAW fence
    FragAB pF[2];
    load_fragA(pF[0], Pw + 0, FLD);
    load_fragA(pF[1], Pw + 32, FLD);

    // O += P V  (B operand = V^T rows, contiguous along k)
#pragma unroll
    for (int dt = 0; dt < 4; ++dt) {
      FragAB vF;
#pragma unroll
      for (int c = 0; c < 2; ++c) {
        load_fragB(vF, &Vs[(dt * 16) * FLD + c * 32], FLD);
        o[dt] = wmma_bf16(pF[c].f, vF.f, o[dt]);
      }
    }
  }

  // normalize and write [B, L, H*64] bf16 for the output projection
#pragma unroll
  for (int v = 0; v < 8; ++v) {
    const float inv = 1.0f / lrow[v];
    const int l = qblk * BQ + wave * 16 + v + hs * 8;
#pragma unroll
    for (int dt = 0; dt < 4; ++dt) {
      const int d = dt * 16 + r15;
      AO[((size_t)(b * LQ + l)) * DM + h * DKH + d] = (__bf16)(o[dt][v] * inv);
    }
  }
}

// ---------------------------------------------------------------------------
extern "C" void kernel_launch(void* const* d_in, const int* in_sizes, int n_in,
                              void* d_out, int out_size, void* d_ws,
                              size_t ws_size, hipStream_t stream) {
  (void)in_sizes; (void)n_in; (void)out_size; (void)ws_size;

  const float* q  = (const float*)d_in[0];
  const float* k  = (const float*)d_in[1];
  const float* v  = (const float*)d_in[2];
  const float* Wq = (const float*)d_in[3];
  const float* bq = (const float*)d_in[4];
  const float* Wk = (const float*)d_in[5];
  const float* bk = (const float*)d_in[6];
  const float* Wv = (const float*)d_in[7];
  const float* bv = (const float*)d_in[8];
  const float* Wo = (const float*)d_in[9];
  const float* bo = (const float*)d_in[10];
  float* out = (float*)d_out;

  const size_t MN = (size_t)MROWS * DM;  // 8.39M elems
  __bf16* Qp = (__bf16*)d_ws;            // [B,H,L,64]
  __bf16* Kp = Qp + MN;                  // [B,H,L,64]
  __bf16* Vt = Kp + MN;                  // [B,H,64,L]
  __bf16* AO = Vt + MN;                  // [B,L,1024]

  dim3 blk(128);
  dim3 gg(MROWS / BM, DM / BN);  // 64 x 16

  gemm_wmma<float, OUT_HEADS><<<gg, blk, 0, stream>>>(q, Wq, bq, (void*)Qp);
  gemm_wmma<float, OUT_HEADS><<<gg, blk, 0, stream>>>(k, Wk, bk, (void*)Kp);
  gemm_wmma<float, OUT_HEADS_T><<<gg, blk, 0, stream>>>(v, Wv, bv, (void*)Vt);

  flash_attn<<<dim3(4 * NH * (LQ / BQ)), blk, 0, stream>>>(Qp, Kp, Vt, AO);

  gemm_wmma<__bf16, OUT_F32><<<gg, blk, 0, stream>>>(AO, Wo, bo, (void*)out);
}
